// SpatialGATAutoEncoder_30940944400706
// MI455X (gfx1250) — matrix operand order repaired and checked by reading, not compiled, not run on hardware
//
#include <hip/hip_runtime.h>
#include <hip/hip_bf16.h>

// ---------------- problem constants (from reference) ----------------
#define N_CELLS    30000
#define N_FEATURES 1000
#define N_EDGES    300000
#define E_TOT      (N_EDGES + N_CELLS)   // real edges + self loops
#define BN_EPS     1e-5f
#define NEG_SLOPE  0.2f

typedef __attribute__((ext_vector_type(16))) __bf16 v16bf;
typedef __attribute__((ext_vector_type(8)))  float  v8f;

// ---------------- helpers ----------------
static __device__ __forceinline__ __bf16 f2bf(float f) {
  // round-to-nearest-even f32 -> bf16 (used only in pack kernels, off the hot path)
  unsigned u = __float_as_uint(f);
  unsigned r = (u + 0x7FFFu + ((u >> 16) & 1u)) >> 16;
  return __builtin_bit_cast(__bf16, (unsigned short)r);
}
static __device__ __forceinline__ float fatomic_add(float* p, float v) {
  // lowers to global_atomic_add_f32 (agent scope, relaxed)
  return __hip_atomic_fetch_add(p, v, __ATOMIC_RELAXED, __HIP_MEMORY_SCOPE_AGENT);
}
// order-preserving float<->uint for atomicMax on floats (incl. negatives)
static __device__ __forceinline__ unsigned fenc(float f) {
  unsigned u = __float_as_uint(f);
  return (u & 0x80000000u) ? ~u : (u | 0x80000000u);
}
static __device__ __forceinline__ float fdec(unsigned e) {
  return (e & 0x80000000u) ? __uint_as_float(e & 0x7FFFFFFFu) : __uint_as_float(~e);
}

// ---------------- pack kernels: f32 row-major -> fragment-major bf16 ----------------
// A-fragment layout (16x32 bf16 tile): lane l holds row mt*16+(l&15);
//   element j maps to K = kc*32 + ((l&16)?8:0) + (j<8 ? j : j+8).
// Stored at Xp[((mt*nkc + kc)*32 + l)*16 + j]   (one v16bf = 32B per lane per tile)
__global__ void pack_a_bf16(const float* __restrict__ X, __bf16* __restrict__ Xp,
                            int Nrows, int Kin)
{
  const int nkc = (Kin + 31) >> 5;
  const int tilesM = Nrows >> 4;
  const long long total = (long long)tilesM * nkc * 512;
  long long i = (long long)blockIdx.x * blockDim.x + threadIdx.x;
  if (i >= total) return;
  const int j  = (int)(i & 15);
  const int l  = (int)((i >> 4) & 31);
  const long long t = i >> 9;
  const int kc = (int)(t % nkc);
  const int mt = (int)(t / nkc);
  const int row = (mt << 4) + (l & 15);
  const int hi8 = (l & 16) ? 8 : 0;
  const int k = (kc << 5) + hi8 + ((j < 8) ? j : (j + 8));
  const float v = (k < Kin) ? X[(size_t)row * Kin + k] : 0.f;
  Xp[i] = f2bf(v);
}

// B-fragment layout (32x16 bf16 tile): lane l holds column tn*16+(l&15);
//   element j maps to K = kc*32 + ((l&16)?8:0) + (j<8 ? j : j+8).
// Stored at Wp[((tn*nkc + kc)*32 + l)*16 + j]
__global__ void pack_w_bf16(const float* __restrict__ W, __bf16* __restrict__ Wp,
                            int Kin, int Kout)
{
  const int nkc = (Kin + 31) >> 5;
  const int tilesN = (Kout + 15) >> 4;
  const long long total = (long long)tilesN * nkc * 512;
  long long i = (long long)blockIdx.x * blockDim.x + threadIdx.x;
  if (i >= total) return;
  const int j  = (int)(i & 15);
  const int l  = (int)((i >> 4) & 31);
  const long long t = i >> 9;
  const int kc = (int)(t % nkc);
  const int tn = (int)(t / nkc);
  const int col = (tn << 4) + (l & 15);
  const int k = (kc << 5) + ((l & 16) ? 8 : 0) + ((j < 8) ? j : (j + 8));
  const float v = (col < Kout && k < Kin) ? W[(size_t)k * Kout + col] : 0.f;
  Wp[i] = f2bf(v);
}

// ---------------- GEMM: Y[N,Kout] = X @ W, pre-packed bf16 fragments ----------------
// One wave computes a 64x16 strip (4 row-tiles x 1 col-tile), 8 waves per block.
// Inner loop: 1 B-frag load + 4 A-frag loads + 4 WMMAs, no branches, EXEC all-ones.
__global__ __launch_bounds__(256) void gemm_wmma_bf16(
    const __bf16* __restrict__ Ap, const __bf16* __restrict__ Bp,
    float* __restrict__ Y, int Nrows, int Kin, int Kout)
{
  const int lane = threadIdx.x & 31;
  const int wave = threadIdx.x >> 5;
  const int tilesM = Nrows >> 4;                  // N_CELLS % 16 == 0
  const int tilesN = (Kout + 15) >> 4;
  const int nkc    = (Kin + 31) >> 5;
  const int mGroups = (tilesM + 3) >> 2;
  int g = blockIdx.x * (blockDim.x >> 5) + wave;
  if (g >= mGroups * tilesN) return;              // wave-uniform exit
  const int tn  = g / mGroups;
  const int mg  = g - tn * mGroups;
  const int mt0 = mg << 2;

  // clamp row-tiles (duplicated compute is discarded at store; keeps loop uniform)
  int mt[4];
  #pragma unroll
  for (int m = 0; m < 4; ++m) {
    int t = mt0 + m;
    mt[m] = (t < tilesM) ? t : (tilesM - 1);
  }

  const v16bf* __restrict__ av = (const v16bf*)Ap;  // one v16bf per (tile, lane)
  const v16bf* __restrict__ bv = (const v16bf*)Bp;

  size_t aIdx[4];
  #pragma unroll
  for (int m = 0; m < 4; ++m) aIdx[m] = ((size_t)mt[m] * nkc) * 32 + lane;
  size_t bIdx = ((size_t)tn * nkc) * 32 + lane;

  v8f acc0 = {}, acc1 = {}, acc2 = {}, acc3 = {};
  for (int kc = 0; kc < nkc; ++kc) {
    const v16bf b  = bv[bIdx];
    const v16bf a0 = av[aIdx[0]];
    const v16bf a1 = av[aIdx[1]];
    const v16bf a2 = av[aIdx[2]];
    const v16bf a3 = av[aIdx[3]];
    acc0 = __builtin_amdgcn_wmma_f32_16x16x32_bf16(false, a0, false, b, (short)0, acc0, false, false);
    acc1 = __builtin_amdgcn_wmma_f32_16x16x32_bf16(false, a1, false, b, (short)0, acc1, false, false);
    acc2 = __builtin_amdgcn_wmma_f32_16x16x32_bf16(false, a2, false, b, (short)0, acc2, false, false);
    acc3 = __builtin_amdgcn_wmma_f32_16x16x32_bf16(false, a3, false, b, (short)0, acc3, false, false);
    bIdx += 32;
    #pragma unroll
    for (int m = 0; m < 4; ++m) aIdx[m] += 32;
  }

  // D (16x16 f32): VGPR r -> row tile*16 + r + 8*(lane>=16), col = tn*16 + (lane&15)
  const int col   = (tn << 4) + (lane & 15);
  const bool colOK = (col < Kout);
  const int rbase = (lane & 16) ? 8 : 0;
  const int nM = (tilesM - mt0 < 4) ? (tilesM - mt0) : 4;   // wave-uniform
  v8f accs[4] = {acc0, acc1, acc2, acc3};
  for (int m = 0; m < nM; ++m) {
    const int orow = ((mt0 + m) << 4) + rbase;
    if (colOK) {
      #pragma unroll
      for (int r = 0; r < 8; ++r)
        Y[(size_t)(orow + r) * Kout + col] = accs[m][r];
    }
  }
}

// ---------------- attention dot products: a_src[n,h], a_dst[n,h] ----------------
__global__ void att_dots(const float* __restrict__ h,
                         const float* __restrict__ att_src, const float* __restrict__ att_dst,
                         float* __restrict__ asrc, float* __restrict__ adst,
                         int Nn, int H, int C)
{
  int i = blockIdx.x * blockDim.x + threadIdx.x;
  if (i >= Nn * H) return;
  int n = i / H, hh = i - n * H;
  const float* hp = h + (size_t)n * H * C + (size_t)hh * C;
  const float* as = att_src + hh * C;
  const float* ad = att_dst + hh * C;
  float s = 0.f, d = 0.f;
  for (int c = 0; c < C; ++c) { float v = hp[c]; s += v * as[c]; d += v * ad[c]; }
  asrc[i] = s; adst[i] = d;
}

// ---------------- edge softmax passes ----------------
__global__ void edge_max(const int* __restrict__ ei,
                         const float* __restrict__ asrc, const float* __restrict__ adst,
                         unsigned* __restrict__ emax, int H)
{
  int i = blockIdx.x * blockDim.x + threadIdx.x;
  if (i >= E_TOT * H) return;
  int e = i / H, hh = i - e * H;
  int s = (e < N_EDGES) ? ei[e] : (e - N_EDGES);
  int d = (e < N_EDGES) ? ei[N_EDGES + e] : (e - N_EDGES);
  float v = asrc[s * H + hh] + adst[d * H + hh];
  v = (v < 0.f) ? NEG_SLOPE * v : v;
  atomicMax(&emax[d * H + hh], fenc(v));
}

__global__ void edge_exp(const int* __restrict__ ei,
                         const float* __restrict__ asrc, const float* __restrict__ adst,
                         const unsigned* __restrict__ emax,
                         float* __restrict__ ee, float* __restrict__ denom, int H)
{
  int i = blockIdx.x * blockDim.x + threadIdx.x;
  if (i >= E_TOT * H) return;
  int e = i / H, hh = i - e * H;
  int s = (e < N_EDGES) ? ei[e] : (e - N_EDGES);
  int d = (e < N_EDGES) ? ei[N_EDGES + e] : (e - N_EDGES);
  float v = asrc[s * H + hh] + adst[d * H + hh];
  v = (v < 0.f) ? NEG_SLOPE * v : v;
  float x = __expf(v - fdec(emax[d * H + hh]));
  ee[i] = x;
  fatomic_add(&denom[d * H + hh], x);
}

__global__ void edge_agg(const int* __restrict__ ei,
                         const float* __restrict__ ee, const float* __restrict__ denom,
                         const float* __restrict__ hfeat, float* __restrict__ agg,
                         int H, int C)
{
  int i = blockIdx.x * blockDim.x + threadIdx.x;
  if (i >= E_TOT * H) return;
  int e = i / H, hh = i - e * H;
  int s = (e < N_EDGES) ? ei[e] : (e - N_EDGES);
  int d = (e < N_EDGES) ? ei[N_EDGES + e] : (e - N_EDGES);
  float alpha = ee[i] / (denom[d * H + hh] + 1e-16f);
  const float* hp = hfeat + (size_t)s * H * C + (size_t)hh * C;
  float*       op = agg   + (size_t)d * H * C + (size_t)hh * C;
  for (int c = 0; c < C; ++c) fatomic_add(&op[c], alpha * hp[c]);
}

// ---------------- elementwise ----------------
__global__ void zero_u32(unsigned* __restrict__ p, long long n) {
  long long i = (long long)blockIdx.x * blockDim.x + threadIdx.x;
  if (i < n) p[i] = 0u;
}
__global__ void bias_add(float* __restrict__ y, const float* __restrict__ bias,
                         long long total, int D) {
  long long i = (long long)blockIdx.x * blockDim.x + threadIdx.x;
  if (i >= total) return;
  y[i] += bias[(int)(i % D)];
}

// ---------------- batch norm ----------------
__global__ __launch_bounds__(256) void bn_stats(const float* __restrict__ x,
                                                float* __restrict__ mean, float* __restrict__ var,
                                                int Nn, int D)
{
  int d = blockIdx.x;
  float s = 0.f, ss = 0.f;
  for (int n = threadIdx.x; n < Nn; n += blockDim.x) {
    float v = x[(size_t)n * D + d];
    s += v; ss += v * v;
  }
  __shared__ float sh[256], sh2[256];
  sh[threadIdx.x] = s; sh2[threadIdx.x] = ss;
  __syncthreads();
  for (int o = blockDim.x >> 1; o > 0; o >>= 1) {
    if ((int)threadIdx.x < o) { sh[threadIdx.x] += sh[threadIdx.x + o]; sh2[threadIdx.x] += sh2[threadIdx.x + o]; }
    __syncthreads();
  }
  if (threadIdx.x == 0) {
    float m = sh[0] / (float)Nn;
    mean[d] = m;
    var[d]  = sh2[0] / (float)Nn - m * m;   // biased variance
  }
}
__global__ void bn_relu(float* __restrict__ y,
                        const float* __restrict__ mean, const float* __restrict__ var,
                        const float* __restrict__ g, const float* __restrict__ b,
                        long long total, int D)
{
  long long i = (long long)blockIdx.x * blockDim.x + threadIdx.x;
  if (i >= total) return;
  int d = (int)(i % D);
  float v = (y[i] - mean[d]) * rsqrtf(var[d] + BN_EPS) * g[d] + b[d];
  y[i] = v > 0.f ? v : 0.f;
}

// ---------------- host-side orchestration ----------------
struct Conv { const float *W, *att_dst, *att_src, *bias; int in, H, C; };

static inline int cdiv(long long a, int b) { return (int)((a + b - 1) / b); }

static void run_conv(const float* xin, float* hbuf, float* agg, const Conv& p, const int* ei,
                     __bf16* xpack, __bf16* wpack,
                     float* asrcb, float* adstb, unsigned* emaxb, float* denomb, float* eeb,
                     hipStream_t s)
{
  const int HC  = p.H * p.C;
  const int nkc = (p.in + 31) >> 5;
  const int tilesM = N_CELLS >> 4;
  const int tilesN = (HC + 15) >> 4;
  // 0) pack A and B into fragment-major bf16 (K zero-padded)
  const long long aElems = (long long)tilesM * nkc * 512;
  const long long wElems = (long long)tilesN * nkc * 512;
  pack_a_bf16<<<cdiv(aElems, 256), 256, 0, s>>>(xin, xpack, N_CELLS, p.in);
  pack_w_bf16<<<cdiv(wElems, 256), 256, 0, s>>>(p.W, wpack, p.in, HC);
  // 1) GEMM  h = x @ W
  const int mGroups = (tilesM + 3) >> 2;
  const int waves = mGroups * tilesN;
  gemm_wmma_bf16<<<cdiv(waves, 8), 256, 0, s>>>(xpack, wpack, hbuf, N_CELLS, p.in, HC);
  // 2) attention dots
  const int nh = N_CELLS * p.H;
  att_dots<<<cdiv(nh, 256), 256, 0, s>>>(hbuf, p.att_src, p.att_dst, asrcb, adstb, N_CELLS, p.H, p.C);
  // 3) clear softmax state + output accumulator
  zero_u32<<<cdiv(nh, 256), 256, 0, s>>>(emaxb, nh);
  zero_u32<<<cdiv(nh, 256), 256, 0, s>>>((unsigned*)denomb, nh);
  const long long aggN = (long long)N_CELLS * HC;
  zero_u32<<<cdiv(aggN, 256), 256, 0, s>>>((unsigned*)agg, aggN);
  // 4) edge softmax + aggregation
  const int eh = E_TOT * p.H;
  edge_max<<<cdiv(eh, 256), 256, 0, s>>>(ei, asrcb, adstb, emaxb, p.H);
  edge_exp<<<cdiv(eh, 256), 256, 0, s>>>(ei, asrcb, adstb, emaxb, eeb, denomb, p.H);
  edge_agg<<<cdiv(eh, 256), 256, 0, s>>>(ei, eeb, denomb, hbuf, agg, p.H, p.C);
  // 5) bias (concat layout == [N,H*C]; non-concat layers all have H==1 so mean is identity)
  bias_add<<<cdiv(aggN, 256), 256, 0, s>>>(agg, p.bias, aggN, HC);
}

static void run_bn_relu(float* y, const float* g, const float* b, int D,
                        float* meanb, float* varb, hipStream_t s)
{
  bn_stats<<<D, 256, 0, s>>>(y, meanb, varb, N_CELLS, D);
  const long long total = (long long)N_CELLS * D;
  bn_relu<<<cdiv(total, 256), 256, 0, s>>>(y, meanb, varb, g, b, total, D);
}

extern "C" void kernel_launch(void* const* d_in, const int* in_sizes, int n_in,
                              void* d_out, int out_size, void* d_ws, size_t ws_size,
                              hipStream_t stream)
{
  (void)in_sizes; (void)n_in; (void)out_size; (void)ws_size;
  const float* x  = (const float*)d_in[0];
  const int*   ei = (const int*)d_in[1];
  auto F = [&](int i) { return (const float*)d_in[i]; };

  // params flattened with sorted dict keys: bn_dec, bn_enc, dec, enc;
  // each conv dict sorted: W, att_dst, att_src, bias
  // bn_dec (dims 64,128,256): idx 2..7 ; bn_enc (dims 256,128,64): idx 8..13
  const Conv dec0{F(14), F(15), F(16), F(17),  32, 4,   16};
  const Conv dec1{F(18), F(19), F(20), F(21),  64, 4,   32};
  const Conv dec2{F(22), F(23), F(24), F(25), 128, 4,   64};
  const Conv dec3{F(26), F(27), F(28), F(29), 256, 1, 1000};
  const Conv enc0{F(30), F(31), F(32), F(33), 1000, 4,  64};
  const Conv enc1{F(34), F(35), F(36), F(37),  256, 4,  32};
  const Conv enc2{F(38), F(39), F(40), F(41),  128, 4,  16};
  const Conv enc3{F(42), F(43), F(44), F(45),   64, 1,  32};

  // ---- carve workspace ----
  char* w = (char*)d_ws;
  auto alloc = [&](size_t bytes) { void* p = (void*)w; w += (bytes + 255) & ~(size_t)255; return p; };
  float*    hbuf   = (float*)   alloc((size_t)N_CELLS * 1024 * 4);      // up to H*C = 1000
  float*    bufA   = (float*)   alloc((size_t)N_CELLS * 256 * 4);
  float*    bufB   = (float*)   alloc((size_t)N_CELLS * 256 * 4);
  __bf16*   xpack  = (__bf16*)  alloc((size_t)(N_CELLS / 16) * 32 * 512 * 2); // tilesM*nkc(max 32)*512 bf16
  __bf16*   wpack  = (__bf16*)  alloc((size_t)1 << 20);                 // <= 0.55 MB needed
  float*    asrcb  = (float*)   alloc((size_t)N_CELLS * 4 * 4);
  float*    adstb  = (float*)   alloc((size_t)N_CELLS * 4 * 4);
  unsigned* emaxb  = (unsigned*)alloc((size_t)N_CELLS * 4 * 4);
  float*    denomb = (float*)   alloc((size_t)N_CELLS * 4 * 4);
  float*    eeb    = (float*)   alloc((size_t)E_TOT * 4 * 4);
  float*    meanb  = (float*)   alloc(1024 * 4);
  float*    varb   = (float*)   alloc(1024 * 4);

  // ---- encoder ----
  run_conv(x,    hbuf, bufA, enc0, ei, xpack, wpack, asrcb, adstb, emaxb, denomb, eeb, stream); // [N,256]
  run_bn_relu(bufA, F(8),  F(9),  256, meanb, varb, stream);
  run_conv(bufA, hbuf, bufB, enc1, ei, xpack, wpack, asrcb, adstb, emaxb, denomb, eeb, stream); // [N,128]
  run_bn_relu(bufB, F(10), F(11), 128, meanb, varb, stream);
  run_conv(bufB, hbuf, bufA, enc2, ei, xpack, wpack, asrcb, adstb, emaxb, denomb, eeb, stream); // [N,64]
  run_bn_relu(bufA, F(12), F(13), 64, meanb, varb, stream);
  run_conv(bufA, hbuf, bufB, enc3, ei, xpack, wpack, asrcb, adstb, emaxb, denomb, eeb, stream); // [N,32] latent
  // ---- decoder ----
  run_conv(bufB, hbuf, bufA, dec0, ei, xpack, wpack, asrcb, adstb, emaxb, denomb, eeb, stream); // [N,64]
  run_bn_relu(bufA, F(2),  F(3),  64, meanb, varb, stream);
  run_conv(bufA, hbuf, bufB, dec1, ei, xpack, wpack, asrcb, adstb, emaxb, denomb, eeb, stream); // [N,128]
  run_bn_relu(bufB, F(4),  F(5),  128, meanb, varb, stream);
  run_conv(bufB, hbuf, bufA, dec2, ei, xpack, wpack, asrcb, adstb, emaxb, denomb, eeb, stream); // [N,256]
  run_bn_relu(bufA, F(6),  F(7),  256, meanb, varb, stream);
  run_conv(bufA, hbuf, (float*)d_out, dec3, ei, xpack, wpack, asrcb, adstb, emaxb, denomb, eeb, stream); // [N,1000]
}
